// SDETransitionTimeIndep_25116968747681
// MI455X (gfx1250) — compile-verified, matrix operand order checked by reading, not run on hardware
//
#include <hip/hip_runtime.h>

// ---------------------------------------------------------------------------
// SDE transition (time-independent drift/diffusion MLPs), MI455X / gfx1250.
// B=65536 rows, D=64, H=256, 10 Euler-Maruyama steps.
// ~258 GFLOP of GEMM vs ~508MB of output traffic => matrix-core bound.
// Strategy: f16 WMMA (v_wmma_f32_16x16x32_f16) with f32 accum, weights packed
// once into WMMA-B fragment layout (L2-resident, 384KB), activations staged in
// LDS, fully fused per-step kernel (6 GEMM layers + pointwise epilogues).
// ---------------------------------------------------------------------------

typedef __attribute__((ext_vector_type(16))) _Float16 v16h;
typedef __attribute__((ext_vector_type(8)))  _Float16 v8h;
typedef __attribute__((ext_vector_type(8)))  float    v8f;

#define DEV __device__ __forceinline__

namespace sde {
constexpr int B     = 65536;
constexpr int D     = 64;
constexpr int H     = 256;
constexpr int STEPS = 10;

constexpr int ROWS  = 64;    // batch rows per workgroup (4 row tiles of 16)
constexpr int NTHR  = 256;   // 8 wave32s
constexpr int ZSTR  = 72;    // LDS stride (halves) for 64-wide z tile  (144B = 16*9, bank-spread)
constexpr int HSTR  = 264;   // LDS stride (halves) for 256-wide tiles  (528B = 16*33, bank-spread)

// packed-weight half-element offsets inside d_ws
constexpr size_t OFF_MU1 = 0;                       // 64x256
constexpr size_t OFF_MU2 = OFF_MU1 + 64 * 256;      // 256x256
constexpr size_t OFF_MU3 = OFF_MU2 + 256 * 256;     // 256x256
constexpr size_t OFF_MU4 = OFF_MU3 + 256 * 256;     // 256x64
constexpr size_t OFF_SG1 = OFF_MU4 + 256 * 64;      // 64x256
constexpr size_t OFF_SG2 = OFF_SG1 + 64 * 256;      // 256x64
}

// ---------------------------------------------------------------------------
// Pack a row-major f32 weight matrix W[K][N] into WMMA-B fragment layout, f16.
// For each 32x16 (KxN) tile, lane l owns column n = tile_n*16 + (l&15) and the
// K-subset {8h..8h+7, 16+8h..16+8h+7} with h = l>>4 (ISA 16-bit B layout, the
// transpose-symmetric image of the documented 16-bit A 16x32 layout).
// Lane data is stored as 16 contiguous halves => the GEMM loads one aligned
// 32-byte v16h per fragment (2x global_load_b128, L2-hit).
// ---------------------------------------------------------------------------
__global__ void sde_pack_weights(const float* __restrict__ W,
                                 _Float16* __restrict__ out, int K, int N) {
    int idx = blockIdx.x * blockDim.x + threadIdx.x;
    if (idx >= K * N) return;
    int k = idx / N, n = idx % N;
    int kt = k >> 5, kl = k & 31;
    int nt = n >> 4, nl = n & 15;
    int hi    = kl >> 4;        // second 16-K half of the tile?
    int rem   = kl & 15;
    int khalf = rem >> 3;       // which lane group (0..15 vs 16..31)
    int jj    = rem & 7;
    int j     = hi * 8 + jj;    // element index inside the lane's v16h
    int lane  = khalf * 16 + nl;
    int numNt = N >> 4;
    out[((size_t)((kt * numNt + nt) * 32 + lane) << 4) + j] = (_Float16)W[idx];
}

// ---------------------------------------------------------------------------
// Wave-level GEMM over a ROWSx(16*numNt) output: for each owned column tile,
// register-cache KT B-fragments, then sweep row tiles accumulating KT wmmas.
// A fragments come from a padded LDS tile (two ds_load_b128 per fragment).
// ---------------------------------------------------------------------------
template <int KT, typename Epi>
DEV void wmma_gemm(const _Float16* __restrict__ aT, int aStride,
                   const _Float16* __restrict__ pw, int numNt,
                   const float* __restrict__ bias,
                   int nt0, int ntStep, int mt0, int mt1, int lane, Epi epi) {
    for (int nt = nt0; nt < numNt; nt += ntStep) {
        v16h bf[KT];
#pragma unroll
        for (int kt = 0; kt < KT; ++kt)
            bf[kt] = *(const v16h*)(pw + ((size_t)((kt * numNt + nt) * 32 + lane) << 4));
        float bv = bias[nt * 16 + (lane & 15)];
        for (int mt = mt0; mt < mt1; ++mt) {
            const _Float16* arow =
                aT + (mt * 16 + (lane & 15)) * aStride + ((lane >> 4) << 3);
            v8f c = {};
#pragma unroll
            for (int kt = 0; kt < KT; ++kt) {
                v8h lo = *(const v8h*)(arow + kt * 32);       // K = k0..k0+7
                v8h hi = *(const v8h*)(arow + kt * 32 + 16);  // K = k0+16..k0+23
                v16h a;
#pragma unroll
                for (int i = 0; i < 8; ++i) { a[i] = lo[i]; a[8 + i] = hi[i]; }
                c = __builtin_amdgcn_wmma_f32_16x16x32_f16(
                        false, a, false, bf[kt], (short)0, c, false, false);
            }
            epi(mt, nt, bv, c);
        }
    }
}

// ---------------------------------------------------------------------------
// One fused SDE step: drift MLP (4 layers), sigma MLP (2 layers + softplus),
// Euler-Maruyama update. One workgroup handles 64 batch rows.
// ---------------------------------------------------------------------------
__global__ __launch_bounds__(sde::NTHR) void sde_step(
    const float* __restrict__ z_in, float* __restrict__ z_out,
    float* __restrict__ drift_out, float* __restrict__ diff_out,
    const float* __restrict__ dt, const float* __restrict__ eps_t,
    const _Float16* __restrict__ pw,
    const float* __restrict__ mu_b1, const float* __restrict__ mu_b2,
    const float* __restrict__ mu_b3, const float* __restrict__ mu_b4,
    const float* __restrict__ sg_b1, const float* __restrict__ sg_b2) {
    using namespace sde;

    __shared__ _Float16 zA[ROWS * ZSTR];     // 9 KB   f16 z tile (shared by both MLPs)
    __shared__ _Float16 act0[ROWS * HSTR];   // 33.8KB activation ping
    __shared__ _Float16 act1[ROWS * HSTR];   // 33.8KB activation pong
    __shared__ float    driftT[ROWS * D];    // 16 KB  f32 drift tile

    const int tid  = threadIdx.x;
    const int lane = tid & 31;
    const int wave = tid >> 5;
    const int rowBase = blockIdx.x * ROWS;

    // stage z tile -> f16 LDS (coalesced)
    for (int e = tid; e < ROWS * D; e += NTHR) {
        int r = e >> 6, c = e & 63;
        zA[r * ZSTR + c] = (_Float16)z_in[(size_t)(rowBase + r) * D + c];
    }
    __syncthreads();

    const _Float16* pw_mu1 = pw + OFF_MU1;
    const _Float16* pw_mu2 = pw + OFF_MU2;
    const _Float16* pw_mu3 = pw + OFF_MU3;
    const _Float16* pw_mu4 = pw + OFF_MU4;
    const _Float16* pw_sg1 = pw + OFF_SG1;
    const _Float16* pw_sg2 = pw + OFF_SG2;

    // epilogues (C/D layout: VGPR r -> M = r + 8*(lane>=16), N = lane&15)
    auto reluAct0 = [&](int mt, int nt, float bv, v8f c) {
        int col = nt * 16 + (lane & 15);
        int rb  = mt * 16 + ((lane >> 4) << 3);
#pragma unroll
        for (int r = 0; r < 8; ++r) {
            float v = c[r] + bv;
            act0[(rb + r) * HSTR + col] = (_Float16)(v > 0.f ? v : 0.f);
        }
    };
    auto reluAct1 = [&](int mt, int nt, float bv, v8f c) {
        int col = nt * 16 + (lane & 15);
        int rb  = mt * 16 + ((lane >> 4) << 3);
#pragma unroll
        for (int r = 0; r < 8; ++r) {
            float v = c[r] + bv;
            act1[(rb + r) * HSTR + col] = (_Float16)(v > 0.f ? v : 0.f);
        }
    };
    auto driftEpi = [&](int mt, int nt, float bv, v8f c) {
        int col = nt * 16 + (lane & 15);
        int rb  = mt * 16 + ((lane >> 4) << 3);
#pragma unroll
        for (int r = 0; r < 8; ++r) {
            float v = c[r] + bv;
            int row = rb + r;
            driftT[row * D + col] = v;
            drift_out[(size_t)(rowBase + row) * D + col] = v;
        }
    };
    auto finalEpi = [&](int mt, int nt, float bv, v8f c) {
        int col = nt * 16 + (lane & 15);
        int rb  = mt * 16 + ((lane >> 4) << 3);
#pragma unroll
        for (int r = 0; r < 8; ++r) {
            int row = rb + r;
            int g   = rowBase + row;
            float x  = c[r] + bv;
            // numerically stable softplus
            float sp = fmaxf(x, 0.f) + log1pf(__expf(-fabsf(x)));
            float df = sp + 0.01f;
            diff_out[(size_t)g * D + col] = df;
            float dtv = dt[g];
            float dw  = eps_t[g] * sqrtf(dtv);
            float zc  = z_in[(size_t)g * D + col];
            z_out[(size_t)g * D + col] =
                zc + driftT[row * D + col] * dtv + df * dw;
        }
    };

    // Layer distribution: H-wide layers -> wave owns column tiles {w, w+8},
    // all 4 row tiles. D-wide layers -> wave owns column tile w&3 and a
    // 2-row-tile half (w>>2). All bounds uniform per wave (EXEC all-ones).
    wmma_gemm<2>(zA,   ZSTR, pw_mu1, 16, mu_b1, wave, 8, 0, 4, lane, reluAct0);
    __syncthreads();
    wmma_gemm<8>(act0, HSTR, pw_mu2, 16, mu_b2, wave, 8, 0, 4, lane, reluAct1);
    __syncthreads();
    wmma_gemm<8>(act1, HSTR, pw_mu3, 16, mu_b3, wave, 8, 0, 4, lane, reluAct0);
    __syncthreads();
    wmma_gemm<8>(act0, HSTR, pw_mu4, 4, mu_b4, wave & 3, 8,
                 (wave >> 2) * 2, (wave >> 2) * 2 + 2, lane, driftEpi);
    __syncthreads();
    wmma_gemm<2>(zA,   ZSTR, pw_sg1, 16, sg_b1, wave, 8, 0, 4, lane, reluAct1);
    __syncthreads();
    wmma_gemm<8>(act1, HSTR, pw_sg2, 4, sg_b2, wave & 3, 8,
                 (wave >> 2) * 2, (wave >> 2) * 2 + 2, lane, finalEpi);
}

// ---------------------------------------------------------------------------
extern "C" void kernel_launch(void* const* d_in, const int* in_sizes, int n_in,
                              void* d_out, int out_size, void* d_ws, size_t ws_size,
                              hipStream_t stream) {
    using namespace sde;
    const float* latents = (const float*)d_in[0];
    // d_in[1] = start_times (unused by outputs)
    const float* dt    = (const float*)d_in[2];
    const float* eps   = (const float*)d_in[3];
    const float* mu_w1 = (const float*)d_in[4];  const float* mu_b1 = (const float*)d_in[5];
    const float* mu_w2 = (const float*)d_in[6];  const float* mu_b2 = (const float*)d_in[7];
    const float* mu_w3 = (const float*)d_in[8];  const float* mu_b3 = (const float*)d_in[9];
    const float* mu_w4 = (const float*)d_in[10]; const float* mu_b4 = (const float*)d_in[11];
    const float* sg_w1 = (const float*)d_in[12]; const float* sg_b1 = (const float*)d_in[13];
    const float* sg_w2 = (const float*)d_in[14]; const float* sg_b2 = (const float*)d_in[15];

    _Float16* pw = (_Float16*)d_ws;  // 384KB of packed f16 weights

    auto pack = [&](const float* W, size_t off, int K, int N) {
        int n = K * N;
        sde_pack_weights<<<(n + 255) / 256, 256, 0, stream>>>(W, pw + off, K, N);
    };
    pack(mu_w1, OFF_MU1, D, H);
    pack(mu_w2, OFF_MU2, H, H);
    pack(mu_w3, OFF_MU3, H, H);
    pack(mu_w4, OFF_MU4, H, D);
    pack(sg_w1, OFF_SG1, D, H);
    pack(sg_w2, OFF_SG2, H, D);

    float* vals   = (float*)d_out;                        // (11, B, D)
    float* drifts = vals + (size_t)(STEPS + 1) * B * D;   // (10, B, D)
    float* diffs  = drifts + (size_t)STEPS * B * D;       // (10, B, D)

    hipMemcpyAsync(vals, latents, (size_t)B * D * sizeof(float),
                   hipMemcpyDeviceToDevice, stream);

    for (int t = 0; t < STEPS; ++t) {
        sde_step<<<B / ROWS, NTHR, 0, stream>>>(
            vals + (size_t)t * B * D, vals + (size_t)(t + 1) * B * D,
            drifts + (size_t)t * B * D, diffs + (size_t)t * B * D,
            dt, eps + (size_t)t * B, pw,
            mu_b1, mu_b2, mu_b3, mu_b4, sg_b1, sg_b2);
    }
}
